// SlotAttention_66623532695899
// MI455X (gfx1250) — compile-verified, matrix operand order checked by reading, not run on hardware
//
#include <hip/hip_runtime.h>
#include <stdint.h>

#define B_   4
#define N_   32768
#define M_   100
#define MP_  112
#define C_   256
#define TN_  64
#define ATTN_SCALE_ 0.0625f   /* 256^-0.5 */
#define EPS_        1e-6f
#define LN_EPS_     1e-5f

typedef __attribute__((ext_vector_type(16))) __bf16 v16bf;
typedef __attribute__((ext_vector_type(8)))  float  v8f;

__device__ __forceinline__ unsigned short f2bf(float f) {
  unsigned u = __float_as_uint(f);
  u += 0x7FFFu + ((u >> 16) & 1u);           // round-to-nearest-even
  return (unsigned short)(u >> 16);
}
__device__ __forceinline__ float bf2f(unsigned short h) {
  return __uint_as_float(((unsigned)h) << 16);
}

// ---------------- LDS layout (bytes), all offsets 16B aligned ----------------
#define WK_OFF  0                     // Wk bf16  [256][264]
#define WK_STR  264
#define Q_OFF   135168                // q  bf16  [112][264]
#define Q_STR   264
#define X_OFF   194304                // xln bf16 [64][264]  / reused: v^T bf16 [256][72]
#define X_STR   264
#define VT_STR  72
#define K_OFF   231168                // k  bf16  [64][264]
#define K_STR   264
#define LG_OFF  264960                // logits f32 [64][132]
#define LG_STR  132
#define AT_OFF  298752                // attn^T bf16 [112][72]
#define AT_STR  72
#define SMEM_BYTES (298752 + MP_ * AT_STR * 2)   // 314880

// Load one WMMA 16-bit operand fragment from LDS (row-major source, rows are
// the "lane dimension" of the operand). Per ISA 16-bit A/B layout:
// lane holds row (lane&15); K slice = kbase + ((lane>>4)<<3) + {0..7, 16..23}.
__device__ __forceinline__ v16bf ld_frag(const __bf16* base, int row,
                                         int stride_h, int kbase, int lane) {
  const __bf16* p = base + (size_t)row * stride_h + kbase + ((lane >> 4) << 3);
  union { uint4 u[2]; v16bf v; } r;
  r.u[0] = *(const uint4*)p;          // K = kb .. kb+7
  r.u[1] = *(const uint4*)(p + 16);   // K = kb+16 .. kb+23
  return r.v;
}

// ============================ prep kernel ====================================
// blocks 0..399   : q[b][m][:] = bf16( scale * (LN(slots[b][m]) @ Wq^T) )
// blocks 400..415 : wkbf = bf16(Wk)
// block  416      : zero q padding rows m in [100,112)
__global__ void __launch_bounds__(256)
slot_attn_prep(const float* __restrict__ slots,
               const float* __restrict__ ln_q_g,
               const float* __restrict__ ln_q_b,
               const float* __restrict__ Wq,
               const float* __restrict__ Wk,
               __bf16* __restrict__ qbf,
               __bf16* __restrict__ wkbf) {
  __shared__ float red[256];
  __shared__ float lnrow[256];
  const int tid = threadIdx.x;
  const int bid = blockIdx.x;
  if (bid < 400) {
    const int b = bid / 100, m = bid % 100;
    const float x = slots[((size_t)b * 100 + m) * 256 + tid];
    red[tid] = x; __syncthreads();
    for (int s = 128; s > 0; s >>= 1) { if (tid < s) red[tid] += red[tid + s]; __syncthreads(); }
    const float mean = red[0] * (1.f / 256.f);
    __syncthreads();
    red[tid] = x * x; __syncthreads();
    for (int s = 128; s > 0; s >>= 1) { if (tid < s) red[tid] += red[tid + s]; __syncthreads(); }
    const float var = red[0] * (1.f / 256.f) - mean * mean;
    const float rs = rsqrtf(var + LN_EPS_);
    lnrow[tid] = (x - mean) * rs * ln_q_g[tid] + ln_q_b[tid];
    __syncthreads();
    float acc = 0.f;
    const float4* wrow = (const float4*)(Wq + (size_t)tid * 256);
    const float4* l4 = (const float4*)lnrow;
#pragma unroll 8
    for (int i = 0; i < 64; ++i) {
      float4 a = l4[i], w = wrow[i];
      acc += a.x * w.x + a.y * w.y + a.z * w.z + a.w * w.w;
    }
    ((unsigned short*)qbf)[((size_t)b * MP_ + m) * 256 + tid] = f2bf(acc * ATTN_SCALE_);
  } else if (bid < 416) {
    const int k = bid - 400;
    for (int idx = tid; idx < 4096; idx += 256) {
      const size_t lin = (size_t)k * 4096 + idx;
      ((unsigned short*)wkbf)[lin] = f2bf(Wk[lin]);
    }
  } else {
    for (int idx = tid; idx < 12288; idx += 256) {
      const int b2 = idx / 3072, rem = idx % 3072;
      const int mm = 100 + rem / 256, cc = rem & 255;
      ((unsigned short*)qbf)[((size_t)b2 * MP_ + mm) * 256 + cc] = 0;
    }
  }
}

// ============================ main fused kernel ==============================
__global__ void __launch_bounds__(256)
slot_attn_main(const float* __restrict__ inputs,
               const float* __restrict__ inputs_k,
               const float* __restrict__ ln_g,
               const float* __restrict__ ln_b,
               const __bf16* __restrict__ qbf,
               const __bf16* __restrict__ wkbf,
               float* __restrict__ Upart,
               float* __restrict__ Spart,
               int chunks) {
  extern __shared__ char smem[];
  const int tid  = threadIdx.x;
  const int lane = tid & 31;
  const int wave = tid >> 5;
  const int b      = blockIdx.x / chunks;
  const int chunk  = blockIdx.x % chunks;
  const int rowsPer = N_ / chunks;
  const int nTiles  = rowsPer / TN_;
  const int nBase   = chunk * rowsPer;

  __bf16* wk_lds = (__bf16*)(smem + WK_OFF);
  __bf16* q_lds  = (__bf16*)(smem + Q_OFF);
  __bf16* x_lds  = (__bf16*)(smem + X_OFF);   // LN(inputs_k) tile
  __bf16* vt_lds = (__bf16*)(smem + X_OFF);   // v^T tile (reuses x region)
  __bf16* k_lds  = (__bf16*)(smem + K_OFF);
  float*  lg_lds = (float*)(smem + LG_OFF);
  __bf16* at_lds = (__bf16*)(smem + AT_OFF);  // attn^T [112][72]

  // stage Wk (bf16) and this batch's q (bf16) into padded LDS
  for (int idx = tid; idx < 256 * 32; idx += 256) {
    const int r = idx >> 5, seg = idx & 31;
    *(uint4*)(wk_lds + r * WK_STR + seg * 8) = *(const uint4*)(wkbf + (size_t)r * 256 + seg * 8);
  }
  const __bf16* qsrc = qbf + (size_t)b * MP_ * 256;
  for (int idx = tid; idx < MP_ * 32; idx += 256) {
    const int r = idx >> 5, seg = idx & 31;
    *(uint4*)(q_lds + r * Q_STR + seg * 8) = *(const uint4*)(qsrc + (size_t)r * 256 + seg * 8);
  }

  const v8f vzero = {0.f, 0.f, 0.f, 0.f, 0.f, 0.f, 0.f, 0.f};
  v8f uacc[14];
#pragma unroll
  for (int i = 0; i < 14; ++i) uacc[i] = vzero;
  float sacc = 0.f;

  __syncthreads();

  for (int t = 0; t < nTiles; ++t) {
    const int n0 = nBase + t * TN_;

    // ---- P0: LayerNorm(inputs_k) 64x256 tile -> x_lds (bf16). 4 thr/row. ----
    {
      const int r = tid >> 2, qq = tid & 3;
      const float* src = inputs_k + ((size_t)b * N_ + n0 + r) * C_ + qq * 64;
      float4 xv[16];
      float s1 = 0.f, s2 = 0.f;
#pragma unroll
      for (int i = 0; i < 16; ++i) {
        xv[i] = ((const float4*)src)[i];
        s1 += xv[i].x + xv[i].y + xv[i].z + xv[i].w;
        s2 += xv[i].x * xv[i].x + xv[i].y * xv[i].y + xv[i].z * xv[i].z + xv[i].w * xv[i].w;
      }
      s1 += __shfl_xor(s1, 1); s1 += __shfl_xor(s1, 2);
      s2 += __shfl_xor(s2, 1); s2 += __shfl_xor(s2, 2);
      const float mean = s1 * (1.f / 256.f);
      const float var  = s2 * (1.f / 256.f) - mean * mean;
      const float rs   = rsqrtf(var + LN_EPS_);
      unsigned* dst = (unsigned*)(x_lds + r * X_STR + qq * 64);
#pragma unroll
      for (int i = 0; i < 16; ++i) {
        const float4 g4 = ((const float4*)(ln_g + qq * 64))[i];
        const float4 b4 = ((const float4*)(ln_b + qq * 64))[i];
        const unsigned short h0 = f2bf((xv[i].x - mean) * rs * g4.x + b4.x);
        const unsigned short h1 = f2bf((xv[i].y - mean) * rs * g4.y + b4.y);
        const unsigned short h2 = f2bf((xv[i].z - mean) * rs * g4.z + b4.z);
        const unsigned short h3 = f2bf((xv[i].w - mean) * rs * g4.w + b4.w);
        dst[i * 2]     = (unsigned)h0 | ((unsigned)h1 << 16);
        dst[i * 2 + 1] = (unsigned)h2 | ((unsigned)h3 << 16);
      }
    }
    __syncthreads();

    // ---- P1: k = xln @ Wk^T   (64x256, K=256). 8 output tiles per wave. ----
#pragma unroll
    for (int i = 0; i < 8; ++i) {
      const int u = wave * 8 + i;
      const int mt = u >> 4, ct = u & 15;
      v8f acc = vzero;
#pragma unroll
      for (int kk = 0; kk < 8; ++kk) {
        v16bf a  = ld_frag(x_lds,  mt * 16 + (lane & 15), X_STR,  kk * 32, lane);
        v16bf bm = ld_frag(wk_lds, ct * 16 + (lane & 15), WK_STR, kk * 32, lane);
        acc = __builtin_amdgcn_wmma_f32_16x16x32_bf16(false, a, false, bm,
                                                      (short)0, acc, false, false);
      }
      const int col = ct * 16 + (lane & 15);
      const int rbase = (lane >> 4) << 3;
#pragma unroll
      for (int r = 0; r < 8; ++r)
        ((unsigned short*)k_lds)[(size_t)(mt * 16 + rbase + r) * K_STR + col] = f2bf(acc[r]);
    }
    __syncthreads();

    // ---- P2: v tile -> v^T bf16 LDS; logits = k @ q^T -> lg_lds (f32) ------
    {
      const int r = tid >> 2, qq = tid & 3;
      const float* src = inputs + ((size_t)b * N_ + n0 + r) * C_ + qq * 64;
#pragma unroll
      for (int i = 0; i < 16; ++i) {
        const float4 xv = ((const float4*)src)[i];
        const int c = qq * 64 + i * 4;
        ((unsigned short*)vt_lds)[(c + 0) * VT_STR + r] = f2bf(xv.x);
        ((unsigned short*)vt_lds)[(c + 1) * VT_STR + r] = f2bf(xv.y);
        ((unsigned short*)vt_lds)[(c + 2) * VT_STR + r] = f2bf(xv.z);
        ((unsigned short*)vt_lds)[(c + 3) * VT_STR + r] = f2bf(xv.w);
      }
    }
#pragma unroll
    for (int j = 0; j < 4; ++j) {
      const int idx = wave + j * 8;     // 28 tiles: mt(0..3) x qt(0..6)
      if (idx < 28) {
        const int mt = idx / 7, qt = idx % 7;
        v8f acc = vzero;
#pragma unroll
        for (int kk = 0; kk < 8; ++kk) {
          v16bf a  = ld_frag(k_lds, mt * 16 + (lane & 15), K_STR, kk * 32, lane);
          v16bf bm = ld_frag(q_lds, qt * 16 + (lane & 15), Q_STR, kk * 32, lane);
          acc = __builtin_amdgcn_wmma_f32_16x16x32_bf16(false, a, false, bm,
                                                        (short)0, acc, false, false);
        }
        const int col = qt * 16 + (lane & 15);
        const int rbase = (lane >> 4) << 3;
#pragma unroll
        for (int r = 0; r < 8; ++r)
          lg_lds[(size_t)(mt * 16 + rbase + r) * LG_STR + col] = acc[r];
      }
    }
    __syncthreads();

    // ---- P3: softmax over M=100 per row, +EPS, store attn^T bf16 -----------
    {
      const int r = tid >> 2, part = tid & 3;   // 4 threads per row, 25 slots each
      float vals[25];
      float mx = -3.4e38f;
#pragma unroll
      for (int j = 0; j < 25; ++j) {
        vals[j] = lg_lds[(size_t)r * LG_STR + part * 25 + j];
        mx = fmaxf(mx, vals[j]);
      }
      mx = fmaxf(mx, __shfl_xor(mx, 1));
      mx = fmaxf(mx, __shfl_xor(mx, 2));
      float sm = 0.f;
#pragma unroll
      for (int j = 0; j < 25; ++j) { vals[j] = __expf(vals[j] - mx); sm += vals[j]; }
      sm += __shfl_xor(sm, 1);
      sm += __shfl_xor(sm, 2);
      const float inv = 1.f / sm;
#pragma unroll
      for (int j = 0; j < 25; ++j) {
        const int m = part * 25 + j;
        ((unsigned short*)at_lds)[(size_t)m * AT_STR + r] = f2bf(vals[j] * inv + EPS_);
      }
      if (part == 3) {
#pragma unroll
        for (int m = M_; m < MP_; ++m)
          ((unsigned short*)at_lds)[(size_t)m * AT_STR + r] = 0;
      }
    }
    __syncthreads();

    // ---- P4: S[m] partial; U += attn^T @ v (14 persistent tiles per wave) --
    if (tid < MP_) {
      float s = 0.f;
      for (int n = 0; n < TN_; ++n)
        s += bf2f(((const unsigned short*)at_lds)[(size_t)tid * AT_STR + n]);
      sacc += s;
    }
#pragma unroll
    for (int i = 0; i < 14; ++i) {
      const int u = wave * 14 + i;      // 112 tiles: mt(0..6) x ct(0..15)
      const int mt = u >> 4, ct = u & 15;
#pragma unroll
      for (int kk = 0; kk < 2; ++kk) {
        v16bf a  = ld_frag(at_lds, mt * 16 + (lane & 15), AT_STR, kk * 32, lane);
        v16bf bm = ld_frag(vt_lds, ct * 16 + (lane & 15), VT_STR, kk * 32, lane);
        uacc[i] = __builtin_amdgcn_wmma_f32_16x16x32_bf16(false, a, false, bm,
                                                          (short)0, uacc[i], false, false);
      }
    }
    __syncthreads();
  }

  // write per-chunk partials (plain stores -> deterministic)
  if (tid < MP_) Spart[((size_t)b * chunks + chunk) * MP_ + tid] = sacc;
  float* ubase = Upart + ((size_t)b * chunks + chunk) * MP_ * C_;
#pragma unroll
  for (int i = 0; i < 14; ++i) {
    const int u = wave * 14 + i;
    const int mt = u >> 4, ct = u & 15;
    const int col = ct * 16 + (lane & 15);
    const int rbase = (lane >> 4) << 3;
#pragma unroll
    for (int r = 0; r < 8; ++r)
      ubase[(size_t)(mt * 16 + rbase + r) * C_ + col] = uacc[i][r];
  }
}

// ============================ reduce kernel ==================================
__global__ void __launch_bounds__(256)
slot_attn_reduce(const float* __restrict__ Upart,
                 const float* __restrict__ Spart,
                 float* __restrict__ out, int chunks) {
  const int idx = blockIdx.x * 256 + threadIdx.x;
  if (idx >= M_ * B_ * C_) return;
  const int m = idx / (B_ * C_);
  const int r = idx % (B_ * C_);
  const int b = r / C_;
  const int c = r % C_;
  float su = 0.f, ss = 0.f;
  for (int ch = 0; ch < chunks; ++ch) {
    const size_t base = ((size_t)b * chunks + ch) * MP_ + m;
    su += Upart[base * C_ + c];
    ss += Spart[base];
  }
  out[idx] = su / ss;   // [M, B, C]
}

// ============================ launch =========================================
extern "C" void kernel_launch(void* const* d_in, const int* in_sizes, int n_in,
                              void* d_out, int out_size, void* d_ws, size_t ws_size,
                              hipStream_t stream) {
  (void)in_sizes; (void)n_in; (void)out_size;
  const float* inputs   = (const float*)d_in[0];
  const float* inputs_k = (const float*)d_in[1];
  const float* slots    = (const float*)d_in[2];
  const float* ln_in_g  = (const float*)d_in[3];
  const float* ln_in_b  = (const float*)d_in[4];
  const float* Wk       = (const float*)d_in[5];
  const float* ln_q_g   = (const float*)d_in[6];
  const float* ln_q_b   = (const float*)d_in[7];
  const float* Wq       = (const float*)d_in[8];
  float* out = (float*)d_out;

  char* ws = (char*)d_ws;
  __bf16* qbf   = (__bf16*)(ws);                 // 4*112*256 bf16 = 229376 B
  __bf16* wkbf  = (__bf16*)(ws + 229376);        // 256*256 bf16   = 131072 B
  float*  Upart = (float*)(ws + 360448);

  const size_t fixed = 360448;
  const size_t perchunk = (size_t)B_ * (MP_ * C_ + MP_) * sizeof(float); // 460544 B
  int chunks = 64;
  while (chunks > 4 && fixed + (size_t)chunks * perchunk > ws_size) chunks >>= 1;
  float* Spart = Upart + (size_t)B_ * chunks * MP_ * C_;

  slot_attn_prep<<<417, 256, 0, stream>>>(slots, ln_q_g, ln_q_b, Wq, Wk, qbf, wkbf);
  slot_attn_main<<<B_ * chunks, 256, SMEM_BYTES, stream>>>(
      inputs, inputs_k, ln_in_g, ln_in_b, qbf, wkbf, Upart, Spart, chunks);
  slot_attn_reduce<<<(M_ * B_ * C_ + 255) / 256, 256, 0, stream>>>(Upart, Spart, out, chunks);
}
